// Layer_21062519620181
// MI455X (gfx1250) — matrix operand order repaired and checked by reading, not compiled, not run on hardware
//
#include <hip/hip_runtime.h>
#include <hip/hip_bf16.h>

typedef __attribute__((ext_vector_type(16))) __bf16 v16bf;
typedef __attribute__((ext_vector_type(8)))  float  v8f;

#define E_EDGES   100000
#define N_TILES   6250
#define LATD      256
#define IRD       240
#define OHD       128

// ---------------- packed-weight tile offsets (units: tiles of 512 bf16 = 1KB) -------------
enum : int {
  T_ROUTER = 0,     // 8x1  = 8
  T_TPWE0  = 8,     // 8 experts * (6x7) = 336
  T_TPWS0  = 344,   // 42
  T_TPWE1  = 386,   // 8 * (3x2) = 48
  T_TPWS1  = 434,   // 6
  T_TPWE2  = 440,   // 8 * (2x1) = 16
  T_TPWS2  = 456,   // 2
  T_PW0    = 458,   // 2x4 = 8
  T_PW1    = 466,   // 1x2 = 2
  T_PW2    = 468,   // 1
  T_EW     = 469,   // 8x7 = 56
  T_M1W1   = 525,   // 10x8 = 80
  T_M1W2   = 605,   // 4x8 = 32
  T_M1W3   = 637,   // 4x16 = 64
  T_M2W1   = 701,   // 12x8 = 96
  T_M2W2   = 797,   // 32
  T_M2W3   = 829,   // 64
  T_OH0    = 893,   // 4x4 = 16
  T_OH1    = 909,   // 4x2 = 8
  T_OH2    = 917,   // 4x1 = 4
  T_TOTAL  = 921    // ~0.92 MB of workspace
};

// ---------------- LDS layout (bytes), one 16-edge tile per wave ---------------------------
constexpr int L_LAT   = 0;                     // 16x256 f32 (persistent: residual)
constexpr int L_H1    = L_LAT  + 16*256*4;     // 16x384 f32 ; aliases xp bf16 [3][16][240]
constexpr int L_ONEH  = L_H1   + 16*384*4;     // 16x128 f32
constexpr int L_GATES = L_ONEH + 16*128*4;     // 16x8  f32
constexpr int L_SCR   = L_GATES+ 16*8*4;       // 16x8  f32 (router scratch)
constexpr int L_GBUF  = L_SCR  + 16*8*4;       // 16x48 f32 (sigmoid gates)
constexpr int L_O0    = L_GBUF + 16*48*4;      // 16x112 f32 (TP l=0 out; reused as ew W)
constexpr int L_SV    = L_O0   + 16*112*4;     // 16x240 f32 (scal|V1|V2 -> new_edge)
constexpr int L_Y0    = L_SV   + 16*240*4;     // 16x64  f32 (post scalars)
constexpr int L_Y12   = L_Y0   + 16*64*4;      // 16x176 f32 (Y1|Y2 -> oh weights)
constexpr int L_H2    = L_Y12  + 16*176*4;     // 16x256 f32 (MLP ping-pong)
constexpr int L_META  = L_H2   + 16*256*4;     // ae/ic/in/cutoff
constexpr int SMEM_BYTES = L_META + 4*16*4;    // 107,776 B

// ---------------- WMMA helpers ------------------------------------------------------------
static __device__ __forceinline__ v8f wmma_bf16(v16bf a, v16bf b, v8f c) {
  return __builtin_amdgcn_wmma_f32_16x16x32_bf16(false, a, false, b, (short)0, c, false, false);
}
static __device__ __forceinline__ float siluf(float x) { return x / (1.0f + __expf(-x)); }

// A-operand (16x32 bf16): lanes 0-15 = rows, halves 0..7 = K kb..kb+7, 8..15 = K kb+16..kb+23
static __device__ __forceinline__ v16bf buildA_f32(const float* __restrict__ A, int Ars,
                                                   int kt, int kmax, int lane) {
  const int row = lane & 15;
  const int kb  = kt*32 + ((lane >> 4) << 3);
  const float* p = A + row * Ars;
  v16bf a;
#pragma unroll
  for (int j = 0; j < 16; ++j) {
    int k = kb + ((j < 8) ? j : (8 + j));
    a[j] = (k < kmax) ? (__bf16)p[k] : (__bf16)0.0f;
  }
  return a;
}
static __device__ __forceinline__ v16bf buildA_f32s(const float* __restrict__ A, int Ars,
                                                    int Acs, int kt, int kmax, int lane) {
  const int row = lane & 15;
  const int kb  = kt*32 + ((lane >> 4) << 3);
  const float* p = A + row * Ars;
  v16bf a;
#pragma unroll
  for (int j = 0; j < 16; ++j) {
    int k = kb + ((j < 8) ? j : (8 + j));
    a[j] = (k < kmax) ? (__bf16)p[k * Acs] : (__bf16)0.0f;
  }
  return a;
}
// TP input: column m of [center|edge|neighbor] block for degree L, channel c
template<int L>
static __device__ __forceinline__ v16bf buildA_tp(const __bf16* __restrict__ xp,
                                                  int kt, int c, int lane) {
  constexpr int seg  = (L==0)?64:((L==1)?32:16);
  constexpr int base = (L==0)?0:((L==1)?64:160);
  constexpr int cd   = 2*L+1;
  constexpr int Min  = seg*3;
  const int row = lane & 15;
  const int kb  = kt*32 + ((lane >> 4) << 3);
  v16bf a;
#pragma unroll
  for (int j = 0; j < 16; ++j) {
    int m = kb + ((j < 8) ? j : (8 + j));
    __bf16 v = (__bf16)0.0f;
    if (m < Min) {
      int s = m / seg, mm = m - s*seg;
      v = xp[(s*16 + row)*IRD + base + mm*cd + c];
    }
    a[j] = v;
  }
  return a;
}

// dense [16 x Kdim] @ packed-B -> Out [16 x NDIM], optional bias + silu.
// B tiles are staged into registers first so each K-chain is one clause of
// global_load_b128 pairs + a single wait + back-to-back v_wmma issue.
template<int KT, int NDIM>
static __device__ __forceinline__ void gemm_t(const float* __restrict__ A, int Ars, int Kdim,
                                              const __bf16* __restrict__ Bp,
                                              const float* __restrict__ bias, int act,
                                              float* __restrict__ Out, int Ors, int lane) {
  v16bf Ar[KT];
#pragma unroll
  for (int kt = 0; kt < KT; ++kt) Ar[kt] = buildA_f32(A, Ars, kt, Kdim, lane);
  const v16bf* __restrict__ B = (const v16bf*)Bp;
  constexpr int Nt = (NDIM + 15) >> 4;
  const int col = lane & 15, rb = (lane >> 4) << 3;
#pragma unroll
  for (int nt = 0; nt < Nt; ++nt) {
    v16bf Br[KT];
#pragma unroll
    for (int kt = 0; kt < KT; ++kt) Br[kt] = B[(nt*KT + kt)*32 + lane];
    v8f acc = {};
#pragma unroll
    for (int kt = 0; kt < KT; ++kt) acc = wmma_bf16(Ar[kt], Br[kt], acc);
    int n = nt*16 + col;
    if (n < NDIM) {
      float bv = bias ? bias[n] : 0.0f;
#pragma unroll
      for (int r = 0; r < 8; ++r) {
        float v = acc[r] + bv;
        if (act) v = siluf(v);
        Out[(rb + r)*Ors + n] = v;
      }
    }
  }
}

// per-channel strided variant (single K tile): used by post-linear l=1,2
template<int NDIM>
static __device__ __forceinline__ void gemm_pc(const float* __restrict__ A, int Ars, int Acs,
                                               int Kdim, const __bf16* __restrict__ Bp,
                                               float* __restrict__ Out, int Ors, int Ocs,
                                               int lane) {
  v16bf a = buildA_f32s(A, Ars, Acs, 0, Kdim, lane);
  const v16bf* __restrict__ B = (const v16bf*)Bp;
  constexpr int Nt = (NDIM + 15) >> 4;
  const int col = lane & 15, rb = (lane >> 4) << 3;
#pragma unroll
  for (int nt = 0; nt < Nt; ++nt) {
    v16bf b = B[nt*32 + lane];
    v8f acc = {};
    acc = wmma_bf16(a, b, acc);
    int n = nt*16 + col;
    if (n < NDIM) {
#pragma unroll
      for (int r = 0; r < 8; ++r) Out[(rb + r)*Ors + n*Ocs] = acc[r];
    }
  }
}

// MoE routed TP for degree L: y = sum_k g_k (xin @ We_k) + xin @ Ws, per channel c
template<int L>
static __device__ __forceinline__ void moe_tp(const __bf16* __restrict__ xp,
                                              const float* __restrict__ gates,
                                              const __bf16* __restrict__ BEp,
                                              const __bf16* __restrict__ BSp,
                                              float* __restrict__ Out, int Ors, int lane) {
  constexpr int seg  = (L==0)?64:((L==1)?32:16);
  constexpr int Min  = seg*3;
  constexpr int Kt   = (Min + 31)/32;          // 6, 3, 2(padded)
  constexpr int Mout = (L==0)?112:((L==1)?32:16);
  constexpr int Nt   = (Mout + 15)/16;         // 7, 2, 1
  constexpr int cd   = 2*L+1;
  constexpr int estride = Kt*Nt*32;            // v16bf per expert
  const v16bf* __restrict__ BE = (const v16bf*)BEp;
  const v16bf* __restrict__ BS = (const v16bf*)BSp;
  const int col = lane & 15, rb = (lane >> 4) << 3;
  for (int c = 0; c < cd; ++c) {
    v16bf A[Kt];
#pragma unroll
    for (int kt = 0; kt < Kt; ++kt) A[kt] = buildA_tp<L>(xp, kt, c, lane);
#pragma unroll
    for (int nt = 0; nt < Nt; ++nt) {
      v8f y = {};                               // shared expert (weight 1)
      {
        v16bf Br[Kt];
#pragma unroll
        for (int kt = 0; kt < Kt; ++kt) Br[kt] = BS[(nt*Kt + kt)*32 + lane];
#pragma unroll
        for (int kt = 0; kt < Kt; ++kt) y = wmma_bf16(A[kt], Br[kt], y);
      }
#pragma unroll
      for (int k = 0; k < 8; ++k) {             // routed experts
        v16bf Br[Kt];
#pragma unroll
        for (int kt = 0; kt < Kt; ++kt)
          Br[kt] = BE[(size_t)k*estride + (nt*Kt + kt)*32 + lane];
        v8f acc = {};
#pragma unroll
        for (int kt = 0; kt < Kt; ++kt) acc = wmma_bf16(A[kt], Br[kt], acc);
#pragma unroll
        for (int r = 0; r < 8; ++r) y[r] += gates[(rb + r)*8 + k] * acc[r];
      }
      int n = nt*16 + col;
      if (n < Mout) {
#pragma unroll
        for (int r = 0; r < 8; ++r) Out[(rb + r)*Ors + n*cd + c] = y[r];
      }
    }
  }
}

// ---------------- weight repack: fp32 [K x N] (strides rs,cs) -> WMMA B layout bf16 -------
__global__ void pack_b_kernel(const float* __restrict__ src, int rs, int cs,
                              int K, int N, int Kt, int batchStride,
                              __bf16* __restrict__ dst) {
  int t = blockIdx.x, b = blockIdx.y, lane = threadIdx.x;
  int kt = t % Kt, nt = t / Kt;
  const float* s = src + (size_t)b * batchStride;
  __bf16* d = dst + ((size_t)b * gridDim.x + t) * 512 + lane * 16;
  int n  = nt*16 + (lane & 15);
  int k0 = kt*32 + ((lane >> 4) << 3);
  for (int j = 0; j < 16; ++j) {
    int k = k0 + ((j < 8) ? j : (8 + j));
    float v = (k < K && n < N) ? s[(size_t)k*rs + (size_t)n*cs] : 0.0f;
    d[j] = (__bf16)v;
  }
}

__global__ void init_out_kernel(const float* __restrict__ latents,
                                const float* __restrict__ wigner,
                                float* __restrict__ out_lat, float* __restrict__ out_wig) {
  size_t i = (size_t)blockIdx.x * blockDim.x + threadIdx.x;
  if (i < (size_t)E_EDGES * LATD) out_lat[i] = latents[i];
  if (i == 0) out_wig[0] = wigner[0];
}

// ---------------- fused edge kernel -------------------------------------------------------
struct KParams {
  const float* latents; const float* node_features; const float* edge_features;
  const int* edge_index; const float* cutoff; const int* active_edges;
  const float* edge_one_hot; const float* mole_globals;
  const float* post_b0; const float* ew_b; const float* ln_g; const float* ln_b;
  const float* m1b0; const float* m1b1; const float* m1b2;
  const float* m2b0; const float* m2b1; const float* m2b2;
  const __bf16* wp; float* out_edge; float* out_lat;
};

__global__ __launch_bounds__(32, 1) void fused_edge_kernel(KParams P) {
  extern __shared__ char smem[];
  float*  lat   = (float*)(smem + L_LAT);
  float*  H1    = (float*)(smem + L_H1);
  __bf16* xp    = (__bf16*)(smem + L_H1);      // aliased: dead before H1 is used
  float*  oneh  = (float*)(smem + L_ONEH);
  float*  gates = (float*)(smem + L_GATES);
  float*  scr   = (float*)(smem + L_SCR);
  float*  gbuf  = (float*)(smem + L_GBUF);
  float*  O0    = (float*)(smem + L_O0);
  float*  SV    = (float*)(smem + L_SV);
  float*  Y0    = (float*)(smem + L_Y0);
  float*  Y12   = (float*)(smem + L_Y12);
  float*  H2    = (float*)(smem + L_H2);
  int*    aeI   = (int*)(smem + L_META);
  int*    icI   = aeI + 16;
  int*    inI   = aeI + 32;
  float*  cutA  = (float*)(aeI + 48);

  const int lane = threadIdx.x;
  const int e0 = blockIdx.x * 16;

  if (lane < 16) {
    int ae = P.active_edges[e0 + lane];
    aeI[lane] = ae;
    icI[lane] = P.edge_index[ae];
    inI[lane] = P.edge_index[E_EDGES + ae];
    cutA[lane] = P.cutoff[ae];
  }
  __syncthreads();

  for (int i = lane; i < 16*LATD; i += 32)
    lat[i] = P.latents[(size_t)aeI[i >> 8]*LATD + (i & 255)];
  for (int i = lane; i < 16*OHD; i += 32)
    oneh[i] = P.edge_one_hot[(size_t)aeI[i >> 7]*OHD + (i & 127)];
  for (int i = lane; i < 16*IRD; i += 32) {
    int row = i / IRD, col = i - row*IRD;
    xp[(0*16 + row)*IRD + col] = (__bf16)P.node_features[(size_t)icI[row]*IRD + col];
    xp[(1*16 + row)*IRD + col] = (__bf16)P.edge_features[(size_t)aeI[row]*IRD + col];
    xp[(2*16 + row)*IRD + col] = (__bf16)P.node_features[(size_t)inI[row]*IRD + col];
  }
  __syncthreads();

  // router + softmax(+ mole_globals)
  gemm_t<8, 8>(lat, LATD, LATD, P.wp + (size_t)T_ROUTER*512, nullptr, 0, scr, 8, lane);
  __syncthreads();
  if (lane < 16) {
    float v[8]; float m = -3.0e38f;
    for (int k = 0; k < 8; ++k) { v[k] = scr[lane*8+k] + P.mole_globals[k]; m = fmaxf(m, v[k]); }
    float s = 0.f;
    for (int k = 0; k < 8; ++k) { v[k] = __expf(v[k]-m); s += v[k]; }
    float inv = 1.f/s;
    for (int k = 0; k < 8; ++k) gates[lane*8+k] = v[k]*inv;
  }
  __syncthreads();

  // MoE routed per-l equivariant linear
  moe_tp<0>(xp, gates, P.wp + (size_t)T_TPWE0*512, P.wp + (size_t)T_TPWS0*512, O0, 112, lane);
  moe_tp<1>(xp, gates, P.wp + (size_t)T_TPWE1*512, P.wp + (size_t)T_TPWS1*512, SV + 64, 240, lane);
  moe_tp<2>(xp, gates, P.wp + (size_t)T_TPWE2*512, P.wp + (size_t)T_TPWS2*512, SV + 160, 240, lane);
  __syncthreads();

  // gate nonlinearity
  for (int i = lane; i < 16*64; i += 32) { int r = i >> 6, n = i & 63; SV[r*240+n] = siluf(O0[r*112+n]); }
  for (int i = lane; i < 16*48; i += 32) { int r = i/48, j = i - r*48; gbuf[i] = 1.f/(1.f + __expf(-O0[r*112+64+j])); }
  __syncthreads();
  for (int i = lane; i < 16*96; i += 32) { int r = i/96, t = i - r*96; SV[r*240+64+t]  *= gbuf[r*48 + t/3]; }
  for (int i = lane; i < 16*80; i += 32) { int r = i/80, t = i - r*80; SV[r*240+160+t] *= gbuf[r*48 + 32 + t/5]; }
  __syncthreads();

  // post-linear (bias on scalars only)
  gemm_t<2, 64>(SV, 240, 64, P.wp + (size_t)T_PW0*512, P.post_b0, 0, Y0, 64, lane);
  for (int c = 0; c < 3; ++c)
    gemm_pc<32>(SV + 64 + c, 240, 3, 32, P.wp + (size_t)T_PW1*512, Y12 + c, 176, 3, lane);
  for (int c = 0; c < 5; ++c)
    gemm_pc<16>(SV + 160 + c, 240, 5, 16, P.wp + (size_t)T_PW2*512, Y12 + 96 + c, 176, 5, lane);
  // edge-weight head: w = lat @ ew_W + b (O0 reused)
  gemm_t<8, 112>(lat, LATD, LATD, P.wp + (size_t)T_EW*512, P.ew_b, 0, O0, 112, lane);
  __syncthreads();

  // new_edge (overwrites scal|V1|V2 region)
  for (int i = lane; i < 16*IRD; i += 32) {
    int r = i/IRD, col = i - r*IRD;
    float v;
    if (col < 64)        v = Y0[r*64 + col] * O0[r*112 + col];
    else if (col < 160)  { int t = col - 64;  v = Y12[r*176 + t]      * O0[r*112 + 64 + t/3]; }
    else                 { int t = col - 160; v = Y12[r*176 + 96 + t] * O0[r*112 + 96 + t/5]; }
    SV[r*240 + col] = v;
  }
  __syncthreads();

  // LayerNorm + concat scalars into H1 (xp is dead now)
  if (lane < 16) {
    const float* lr = lat + lane*LATD;
    float mu = 0.f;
    for (int j = 0; j < LATD; ++j) mu += lr[j];
    mu *= (1.f/LATD);
    float var = 0.f;
    for (int j = 0; j < LATD; ++j) { float d = lr[j]-mu; var += d*d; }
    var *= (1.f/LATD);
    float inv = rsqrtf(var + 1e-5f);
    float* hr = H1 + lane*320;
    for (int j = 0; j < LATD; ++j) hr[j] = (lr[j]-mu)*inv*P.ln_g[j] + P.ln_b[j];
    for (int j = 0; j < 64; ++j) hr[256+j] = Y0[lane*64 + j];
  }
  __syncthreads();

  gemm_t<10,128>(H1, 320, 320, P.wp + (size_t)T_M1W1*512, P.m1b0, 1, H2, 256, lane); __syncthreads();
  gemm_t<4, 128>(H2, 256, 128, P.wp + (size_t)T_M1W2*512, P.m1b1, 1, H1, 384, lane); __syncthreads();
  gemm_t<4, 256>(H1, 384, 128, P.wp + (size_t)T_M1W3*512, P.m1b2, 0, H2, 256, lane); __syncthreads();

  for (int i = lane; i < 16*256; i += 32) { int r = i >> 8, cn = i & 255; H1[r*384 + cn] = H2[i]; }
  for (int i = lane; i < 16*128; i += 32) { int r = i >> 7, cn = i & 127; H1[r*384 + 256 + cn] = oneh[i]; }
  __syncthreads();
  gemm_t<12,128>(H1, 384, 384, P.wp + (size_t)T_M2W1*512, P.m2b0, 1, H2, 256, lane); __syncthreads();
  gemm_t<4, 128>(H2, 256, 128, P.wp + (size_t)T_M2W2*512, P.m2b1, 1, H1, 384, lane); __syncthreads();
  gemm_t<4, 256>(H1, 384, 128, P.wp + (size_t)T_M2W3*512, P.m2b2, 0, H2, 256, lane); __syncthreads();

  const float C_OLD = 0.8944271909999159f;   // 1/sqrt(1.25)
  const float C_NEW = 0.4472135954999579f;   // 0.5 * C_OLD
  for (int i = lane; i < 16*LATD; i += 32) {
    int r = i >> 8, cn = i & 255;
    float nl = cutA[r] * H2[i];
    P.out_lat[(size_t)aeI[r]*LATD + cn] = C_NEW*nl + C_OLD*lat[i];
  }

  // one-hot 'uvu' tuning weights (Y12 reused)
  float* wl = Y12;
  gemm_t<4, 64>(oneh, OHD, OHD, P.wp + (size_t)T_OH0*512, nullptr, 0, wl,      112, lane);
  gemm_t<4, 32>(oneh, OHD, OHD, P.wp + (size_t)T_OH1*512, nullptr, 0, wl + 64, 112, lane);
  gemm_t<4, 16>(oneh, OHD, OHD, P.wp + (size_t)T_OH2*512, nullptr, 0, wl + 96, 112, lane);
  __syncthreads();
  const float INV_SQRT_OH = 0.08838834764831845f;  // 1/sqrt(128)
  for (int i = lane; i < 16*IRD; i += 32) {
    int r = i/IRD, col = i - r*IRD;
    int ae = aeI[r];
    float eo = C_OLD*P.edge_features[(size_t)ae*IRD + col] + C_NEW*SV[r*240 + col];
    int widx = (col < 64) ? col : (col < 160 ? 64 + (col-64)/3 : 96 + (col-160)/5);
    eo *= (1.f + wl[r*112 + widx]*INV_SQRT_OH);
    P.out_edge[(size_t)ae*IRD + col] = eo;
  }
}

// ---------------- host launcher -----------------------------------------------------------
extern "C" void kernel_launch(void* const* d_in, const int* in_sizes, int n_in,
                              void* d_out, int out_size, void* d_ws, size_t ws_size,
                              hipStream_t stream) {
  (void)in_sizes; (void)n_in; (void)out_size; (void)ws_size;
  const float* latents      = (const float*)d_in[0];
  const float* node_feat    = (const float*)d_in[1];
  const float* edge_feat    = (const float*)d_in[3];
  const int*   edge_index   = (const int*)d_in[4];
  const float* cutoff       = (const float*)d_in[6];
  const int*   active_edges = (const int*)d_in[7];
  const float* one_hot      = (const float*)d_in[8];
  const float* wigner       = (const float*)d_in[9];
  const float* mole         = (const float*)d_in[10];
  const float* router_W = (const float*)d_in[11];
  const float* tpWe0 = (const float*)d_in[12];
  const float* tpWs0 = (const float*)d_in[13];
  const float* pW0   = (const float*)d_in[14];
  const float* ohW0  = (const float*)d_in[15];
  const float* tpWe1 = (const float*)d_in[16];
  const float* tpWs1 = (const float*)d_in[17];
  const float* pW1   = (const float*)d_in[18];
  const float* ohW1  = (const float*)d_in[19];
  const float* tpWe2 = (const float*)d_in[20];
  const float* tpWs2 = (const float*)d_in[21];
  const float* pW2   = (const float*)d_in[22];
  const float* ohW2  = (const float*)d_in[23];
  const float* post_b0 = (const float*)d_in[24];
  const float* ewW   = (const float*)d_in[25];
  const float* ew_b  = (const float*)d_in[26];
  const float* ln_g  = (const float*)d_in[27];
  const float* ln_b  = (const float*)d_in[28];
  const float* m1W1 = (const float*)d_in[29];
  const float* m1W2 = (const float*)d_in[30];
  const float* m1W3 = (const float*)d_in[31];
  const float* m1b1 = (const float*)d_in[32];
  const float* m1b2 = (const float*)d_in[33];
  const float* m1b3 = (const float*)d_in[34];
  const float* m2W1 = (const float*)d_in[35];
  const float* m2W2 = (const float*)d_in[36];
  const float* m2W3 = (const float*)d_in[37];
  const float* m2b1 = (const float*)d_in[38];
  const float* m2b2 = (const float*)d_in[39];
  const float* m2b3 = (const float*)d_in[40];

  __bf16* wp      = (__bf16*)d_ws;
  float*  out_edge = (float*)d_out;
  float*  out_lat  = out_edge + (size_t)E_EDGES * IRD;
  float*  out_wig  = out_lat  + (size_t)E_EDGES * LATD;

  auto pack = [&](const float* src, int rs, int cs, int K, int N,
                  int Kt, int Nt, int nb, int bstride, int tileOff) {
    pack_b_kernel<<<dim3((unsigned)(Kt*Nt), (unsigned)nb), dim3(32), 0, stream>>>(
        src, rs, cs, K, N, Kt, bstride, wp + (size_t)tileOff * 512);
  };
  pack(router_W,   8, 1, 256,   8,  8, 1, 1, 0,       T_ROUTER);
  pack(tpWe0,    112, 1, 192, 112,  6, 7, 8, 192*112, T_TPWE0);
  pack(tpWs0,    112, 1, 192, 112,  6, 7, 1, 0,       T_TPWS0);
  pack(tpWe1,     32, 1,  96,  32,  3, 2, 8,  96*32,  T_TPWE1);
  pack(tpWs1,     32, 1,  96,  32,  3, 2, 1, 0,       T_TPWS1);
  pack(tpWe2,     16, 1,  48,  16,  2, 1, 8,  48*16,  T_TPWE2);
  pack(tpWs2,     16, 1,  48,  16,  2, 1, 1, 0,       T_TPWS2);
  pack(pW0,       64, 1,  64,  64,  2, 4, 1, 0,       T_PW0);
  pack(pW1,       32, 1,  32,  32,  1, 2, 1, 0,       T_PW1);
  pack(pW2,       16, 1,  16,  16,  1, 1, 1, 0,       T_PW2);
  pack(ewW,      112, 1, 256, 112,  8, 7, 1, 0,       T_EW);
  pack(m1W1,     128, 1, 320, 128, 10, 8, 1, 0,       T_M1W1);
  pack(m1W2,     128, 1, 128, 128,  4, 8, 1, 0,       T_M1W2);
  pack(m1W3,     256, 1, 128, 256,  4,16, 1, 0,       T_M1W3);
  pack(m2W1,     128, 1, 384, 128, 12, 8, 1, 0,       T_M2W1);
  pack(m2W2,     128, 1, 128, 128,  4, 8, 1, 0,       T_M2W2);
  pack(m2W3,     256, 1, 128, 256,  4,16, 1, 0,       T_M2W3);
  // transposed one-hot weights: B[k,n] = ohW[n,k]
  pack(ohW0,       1, 128, 128, 64, 4, 4, 1, 0,       T_OH0);
  pack(ohW1,       1, 128, 128, 32, 4, 2, 1, 0,       T_OH1);
  pack(ohW2,       1, 128, 128, 16, 4, 1, 1, 0,       T_OH2);

  init_out_kernel<<<dim3((E_EDGES*LATD + 255)/256), dim3(256), 0, stream>>>(
      latents, wigner, out_lat, out_wig);

  KParams P;
  P.latents = latents; P.node_features = node_feat; P.edge_features = edge_feat;
  P.edge_index = edge_index; P.cutoff = cutoff; P.active_edges = active_edges;
  P.edge_one_hot = one_hot; P.mole_globals = mole;
  P.post_b0 = post_b0; P.ew_b = ew_b; P.ln_g = ln_g; P.ln_b = ln_b;
  P.m1b0 = m1b1; P.m1b1 = m1b2; P.m1b2 = m1b3;
  P.m2b0 = m2b1; P.m2b1 = m2b2; P.m2b2 = m2b3;
  P.wp = wp; P.out_edge = out_edge; P.out_lat = out_lat;

  (void)hipFuncSetAttribute((const void*)fused_edge_kernel,
                            hipFuncAttributeMaxDynamicSharedMemorySize, SMEM_BYTES);
  fused_edge_kernel<<<dim3(N_TILES), dim3(32), SMEM_BYTES, stream>>>(P);
}
